// HierarchicalMoEFFN_70866960384585
// MI455X (gfx1250) — matrix operand order repaired and checked by reading, not compiled, not run on hardware
//
#include <hip/hip_runtime.h>
#include <hip/hip_bf16.h>
#include <math.h>

// ---------------- constants ----------------
static constexpr int DIM   = 128;   // D
static constexpr int DFF   = 256;
static constexpr int NHH   = 4;
static constexpr int HDIM  = 32;    // head dim
static constexpr int BATCH = 8;
static constexpr int TSEQ  = 1024;
static constexpr int NTOK  = BATCH * TSEQ;  // 8192
static constexpr float BW      = 0.5f / 3.0f;
static constexpr float LB_W    = 0.01f;

typedef _Float16 h16;
typedef __attribute__((ext_vector_type(16))) _Float16 v16h;
typedef __attribute__((ext_vector_type(8)))  _Float16 v8h;
typedef __attribute__((ext_vector_type(8)))  float    v8f;
typedef __attribute__((ext_vector_type(4)))  unsigned int u32x4;
typedef __attribute__((ext_vector_type(8)))  int i32x8;
typedef __attribute__((ext_vector_type(4)))  int i32x4;

#ifndef __has_builtin
#define __has_builtin(x) 0
#endif
#if defined(__HIP_DEVICE_COMPILE__) && \
    __has_builtin(__builtin_amdgcn_tensor_load_to_lds) && \
    __has_builtin(__builtin_amdgcn_s_wait_tensorcnt)
#define USE_TDM 1
#else
#define USE_TDM 0
#endif

// ---------------- wave reductions (wave32) ----------------
__device__ __forceinline__ float wave_red_sum(float v) {
#pragma unroll
  for (int m = 16; m >= 1; m >>= 1) v += __shfl_xor(v, m, 32);
  return v;
}
__device__ __forceinline__ float wave_red_max(float v) {
#pragma unroll
  for (int m = 16; m >= 1; m >>= 1) v = fmaxf(v, __shfl_xor(v, m, 32));
  return v;
}
__device__ __forceinline__ float half_red_sum(float v) {
#pragma unroll
  for (int m = 8; m >= 1; m >>= 1) v += __shfl_xor(v, m, 32);
  return v;
}
__device__ __forceinline__ float half_red_max(float v) {
#pragma unroll
  for (int m = 8; m >= 1; m >>= 1) v = fmaxf(v, __shfl_xor(v, m, 32));
  return v;
}

// ---------------- WMMA fragment helpers ----------------
// A fragment: 16x32 f16, row-major source A[row][k], lda in elements.
__device__ __forceinline__ v16h load_a_frag(const h16* A, int lda, int row0, int k0, int lane) {
  int r  = row0 + (lane & 15);
  int kb = k0 + ((lane >> 4) << 3);
  const h16* p = A + (size_t)r * lda + kb;
  union { v16h v; v8h h[2]; } u;
  u.h[0] = *(const v8h*)(p);
  u.h[1] = *(const v8h*)(p + 16);
  return u.v;
}
// B fragment: 32x16 f16 for (x @ W^T) with row-major W[N][K], ldb in elements.
__device__ __forceinline__ v16h load_b_frag(const h16* W, int ldb, int n0, int k0, int lane) {
  int n  = n0 + (lane & 15);
  int kb = k0 + ((lane >> 4) << 3);
  const h16* p = W + (size_t)n * ldb + kb;
  union { v16h v; v8h h[2]; } u;
  u.h[0] = *(const v8h*)(p);
  u.h[1] = *(const v8h*)(p + 16);
  return u.v;
}
__device__ __forceinline__ v8f wmma16(v16h a, v16h b, v8f c) {
  return __builtin_amdgcn_wmma_f32_16x16x32_f16(false, a, false, b, (short)0, c, false, false);
}

#if USE_TDM
// TDM: async DMA of a contiguous 64KB block (global -> LDS), 1-D descriptor,
// data_size = 8 bytes, 8192 units. Issued by one wave; tracked by TENSORcnt.
__device__ __forceinline__ void tdm_load_64k(const h16* gsrc, h16* lds_dst) {
  uint64_t ga = (uint64_t)(uintptr_t)gsrc;
  uint32_t la = (uint32_t)(uintptr_t)lds_dst;     // low 32 bits = LDS offset
  const unsigned units = (64u * 1024u) / 8u;      // 8192
  u32x4 g0;
  g0[0] = 1u;                                     // count=1 valid user D#
  g0[1] = la;                                     // lds_addr (bytes)
  g0[2] = (unsigned)ga;                           // global_addr[31:0]
  g0[3] = (unsigned)((ga >> 32) & 0x01FFFFFFu) | (2u << 30);  // addr[56:32] | type=2
  i32x8 g1;
  g1[0] = (int)(3u << 16);                        // workgroup_mask=0, data_size=3 (8B)
  g1[1] = (int)((units & 0xFFFFu) << 16);         // tensor_dim0[15:0] @ bits 63:48
  g1[2] = (int)(((units >> 16) & 0xFFFFu) | (1u << 16)); // tensor_dim0 hi | tensor_dim1=1
  g1[3] = (int)((units & 0xFFFFu) << 16);         // tensor_dim1 hi=0 | tile_dim0=units
  g1[4] = 1;                                      // tile_dim1=1, tile_dim2=0
  g1[5] = (int)units;                             // tensor_dim0_stride lo32
  g1[6] = (int)((units & 0xFFFFu) << 16);         // stride hi=0 | dim1_stride lo16
  g1[7] = 0;                                      // dim1_stride hi
  i32x4 z4 = {0, 0, 0, 0};
  i32x8 z8 = {0, 0, 0, 0, 0, 0, 0, 0};
  __builtin_amdgcn_tensor_load_to_lds(g0, g1, z4, z4, z8, 0);
}
#endif

// ---------------- K1: routing / norms / gates ----------------
__global__ __launch_bounds__(256) void routing_kernel(
    const float* __restrict__ x, const float* gr_g, const float* gr_b,
    const float* q6_w, const float* log_temp,
    const float* grp_g, const float* grp_b,
    const float* grp_pw, const float* grp_pb,
    h16* __restrict__ xn_out, float* __restrict__ coef, float* lb_acc)
{
  int wave = (blockIdx.x * blockDim.x + threadIdx.x) >> 5;
  int lane = threadIdx.x & 31;
  if (wave >= NTOK) return;
  const float* xr = x + (size_t)wave * DIM;
  float xv[4]; float s = 0.f;
#pragma unroll
  for (int i = 0; i < 4; ++i) { xv[i] = xr[lane + 32 * i]; s += xv[i]; }
  float mean = wave_red_sum(s) * (1.0f / DIM);
  float vs = 0.f;
#pragma unroll
  for (int i = 0; i < 4; ++i) { float d = xv[i] - mean; vs += d * d; }
  float var = wave_red_sum(vs) * (1.0f / DIM);
  float rs = rsqrtf(var + 1e-5f);
#pragma unroll
  for (int i = 0; i < 4; ++i) {
    xv[i] = (xv[i] - mean) * rs;
    xn_out[(size_t)wave * DIM + lane + 32 * i] = (h16)xv[i];
  }
  float sb[6];
#pragma unroll
  for (int j = 0; j < 6; ++j) {
    float p = 0.f;
#pragma unroll
    for (int i = 0; i < 4; ++i) {
      int d = lane + 32 * i;
      p += (xv[i] * gr_g[d] + gr_b[d]) * q6_w[j * DIM + d];
    }
    sb[j] = tanhf(wave_red_sum(p));
  }
  float temp = fminf(fmaxf(__expf(log_temp[0]), 0.1f), 5.0f);
  float inv_t = 1.0f / temp;
  float sim0 = 0.f, sim1 = 0.f;
  int c0 = lane, c1 = lane + 32;
#pragma unroll
  for (int j = 0; j < 6; ++j) {
    sim0 += sb[j] * (((c0 >> (5 - j)) & 1) ? 1.f : -1.f);
    sim1 += sb[j] * (((c1 >> (5 - j)) & 1) ? 1.f : -1.f);
  }
  sim0 *= inv_t; sim1 *= inv_t;
  float mx = wave_red_max(fmaxf(sim0, sim1));
  float e0 = __expf(sim0 - mx), e1 = __expf(sim1 - mx);
  float inv_sum = 1.0f / wave_red_sum(e0 + e1);
  float sh[6];
#pragma unroll
  for (int j = 0; j < 6; ++j) {
    float p = e0 * (((c0 >> (5 - j)) & 1) ? 1.f : -1.f)
            + e1 * (((c1 >> (5 - j)) & 1) ? 1.f : -1.f);
    sh[j] = wave_red_sum(p) * inv_sum;
  }
  const int pa[3] = {63, 6, 0}, pb[3] = {27, 45, 18};
  float gl[3];
#pragma unroll
  for (int g = 0; g < 3; ++g) {
    float p = 0.f;
#pragma unroll
    for (int j = 0; j < 6; ++j) {
      float av = ((((pa[g] >> (5 - j)) & 1) ? 1.f : -1.f) +
                  (((pb[g] >> (5 - j)) & 1) ? 1.f : -1.f)) * 0.5f;
      p += sh[j] * av;
    }
    gl[g] = p;
  }
  float gm = fmaxf(gl[0], fmaxf(gl[1], gl[2]));
  float ge0 = __expf(gl[0] - gm), ge1 = __expf(gl[1] - gm), ge2 = __expf(gl[2] - gm);
  float gs = 1.0f / (ge0 + ge1 + ge2);
  float gw3[3] = { ge0 * gs, ge1 * gs, ge2 * gs };
#pragma unroll
  for (int g = 0; g < 3; ++g) {
    float p0 = 0.f, p1 = 0.f;
#pragma unroll
    for (int i = 0; i < 4; ++i) {
      int d = lane + 32 * i;
      float hg = xv[i] * grp_g[g * DIM + d] + grp_b[g * DIM + d];
      p0 += hg * grp_pw[(g * 2 + 0) * DIM + d];
      p1 += hg * grp_pw[(g * 2 + 1) * DIM + d];
    }
    float l0 = wave_red_sum(p0) + grp_pb[g * 2 + 0];
    float l1 = wave_red_sum(p1) + grp_pb[g * 2 + 1];
    float m2 = fmaxf(l0, l1);
    float ea = __expf(l0 - m2), eb = __expf(l1 - m2);
    float ssum = ea + eb;
    float w0 = ea / ssum, w1 = eb / ssum;
    float renorm = 1.0f / (w0 + w1 + 1e-8f);
    if (lane == 0) {
      coef[wave * 6 + 2 * g + 0] = gw3[g] * w0 * renorm;
      coef[wave * 6 + 2 * g + 1] = gw3[g] * w1 * renorm;
      atomicAdd(&lb_acc[g], gw3[g]);
      atomicAdd(&lb_acc[3 + 2 * g + 0], w0);
      atomicAdd(&lb_acc[3 + 2 * g + 1], w1);
    }
  }
}

// ---------------- weight prep ----------------
__global__ void cast_f16_kernel(const float* __restrict__ src, h16* __restrict__ dst, int n) {
  for (int i = blockIdx.x * blockDim.x + threadIdx.x; i < n; i += gridDim.x * blockDim.x)
    dst[i] = (h16)src[i];
}
__global__ void expert_fold_kernel(const float* gw, const float* vw,
                                   const float* ng, const float* nb,
                                   h16* Wg, h16* Wv, float* bg, float* bv)
{
  int idx = blockIdx.x * blockDim.x + threadIdx.x;
  if (idx >= 6 * DFF) return;
  int e = idx / DFF, f = idx % DFF;
  const float* grow = gw + (size_t)(e * DFF + f) * DIM;
  const float* vrow = vw + (size_t)(e * DFF + f) * DIM;
  const float* gge = ng + e * DIM;
  const float* bbe = nb + e * DIM;
  float sg = 0.f, sv = 0.f;
  for (int d = 0; d < DIM; ++d) {
    float gval = grow[d], vval = vrow[d];
    Wg[(size_t)(e * DFF + f) * DIM + d] = (h16)(gval * gge[d]);
    Wv[(size_t)(e * DFF + f) * DIM + d] = (h16)(vval * gge[d]);
    sg += bbe[d] * gval; sv += bbe[d] * vval;
  }
  bg[e * DFF + f] = sg; bv[e * DFF + f] = sv;
}

// ---------------- layernorm -> f16 ----------------
__global__ __launch_bounds__(256) void ln_f16_kernel(
    const float* __restrict__ in, const float* g, const float* b,
    h16* __restrict__ out, int rows)
{
  int wave = (blockIdx.x * blockDim.x + threadIdx.x) >> 5;
  int lane = threadIdx.x & 31;
  if (wave >= rows) return;
  const float* r = in + (size_t)wave * DIM;
  float v[4]; float s = 0.f;
#pragma unroll
  for (int i = 0; i < 4; ++i) { v[i] = r[lane + 32 * i]; s += v[i]; }
  float mean = wave_red_sum(s) * (1.0f / DIM);
  float vs = 0.f;
#pragma unroll
  for (int i = 0; i < 4; ++i) { float d = v[i] - mean; vs += d * d; }
  float rs = rsqrtf(wave_red_sum(vs) * (1.0f / DIM) + 1e-5f);
#pragma unroll
  for (int i = 0; i < 4; ++i) {
    int d = lane + 32 * i;
    out[(size_t)wave * DIM + d] = (h16)((v[i] - mean) * rs * g[d] + b[d]);
  }
}

// ---------------- expert fused gated FFN (TDM-staged out weights) ----------------
__global__ __launch_bounds__(128) void expert_ffn_kernel(
    const h16* __restrict__ xn, const h16* __restrict__ wg, const h16* __restrict__ wv,
    const float* __restrict__ bg, const float* __restrict__ bv,
    const h16* __restrict__ wo, const float* __restrict__ coef,
    float* __restrict__ group_outs)
{
  __shared__ __align__(64) h16 gv[16 * DFF];        // 8 KB
  __shared__ __align__(64) h16 wo_lds[DIM * DFF];   // 64 KB
  int wid = threadIdx.x >> 5, lane = threadIdx.x & 31;
  int g = blockIdx.y;
  int row0 = blockIdx.x * 16;
  for (int sub = 0; sub < 2; ++sub) {
    int e = 2 * g + sub;
    const h16* Wg = wg + (size_t)e * DFF * DIM;
    const h16* Wv = wv + (size_t)e * DFF * DIM;
    const h16* Wo = wo + (size_t)e * DIM * DFF;
#if USE_TDM
    if (threadIdx.x < 32) tdm_load_64k(Wo, wo_lds);   // DMA overlaps gate/val GEMMs
#endif
    for (int ft = wid * 4; ft < wid * 4 + 4; ++ft) {
      int n0 = ft * 16;
      v8f ag = {}; v8f av = {};
#pragma unroll
      for (int k0 = 0; k0 < DIM; k0 += 32) {
        v16h a = load_a_frag(xn, DIM, row0, k0, lane);
        ag = wmma16(a, load_b_frag(Wg, DIM, n0, k0, lane), ag);
        av = wmma16(a, load_b_frag(Wv, DIM, n0, k0, lane), av);
      }
      int n = n0 + (lane & 15);
      float bgn = bg[e * DFF + n], bvn = bv[e * DFF + n];
      int mb = (lane >> 4) << 3;
#pragma unroll
      for (int j = 0; j < 8; ++j) {
        float gval = ag[j] + bgn;
        float vval = av[j] + bvn;
        float silu = gval / (1.0f + __expf(-gval));
        gv[(mb + j) * DFF + n] = (h16)(silu * vval);
      }
    }
#if USE_TDM
    if (threadIdx.x < 32) __builtin_amdgcn_s_wait_tensorcnt(0);
#endif
    __syncthreads();
#if USE_TDM
    const h16* Bsrc = wo_lds;
#else
    const h16* Bsrc = Wo;
#endif
    for (int nt = wid * 2; nt < wid * 2 + 2; ++nt) {
      int n0 = nt * 16;
      v8f acc = {};
#pragma unroll
      for (int k0 = 0; k0 < DFF; k0 += 32)
        acc = wmma16(load_a_frag(gv, DFF, 0, k0, lane),
                     load_b_frag(Bsrc, DFF, n0, k0, lane), acc);
      int n = n0 + (lane & 15);
      int mb = (lane >> 4) << 3;
#pragma unroll
      for (int j = 0; j < 8; ++j) {
        int tok = row0 + mb + j;
        float c = coef[tok * 6 + e];
        float* go = group_outs + ((size_t)g * NTOK + tok) * DIM + n;
        float valr = c * acc[j];
        if (sub == 0) *go = valr; else *go += valr;
      }
    }
    __syncthreads();
  }
}

// ---------------- combined init ----------------
__global__ void combine_init_kernel(const float* __restrict__ go, float* __restrict__ combined) {
  int i = blockIdx.x * blockDim.x + threadIdx.x;
  if (i < NTOK * DIM)
    combined[i] = go[i] + go[(size_t)NTOK * DIM + i] + go[2 * (size_t)NTOK * DIM + i];
}

// ---------------- fused QKV projection (2 M-tiles per wave) ----------------
__global__ __launch_bounds__(128) void qkv_kernel(
    const h16* __restrict__ qn, const h16* __restrict__ kvn,
    const h16* __restrict__ w_in,   // [384][128] f16
    h16* __restrict__ Qb, h16* __restrict__ Kb, h16* __restrict__ Vt)
{
  int wid = threadIdx.x >> 5, lane = threadIdx.x & 31;
  int row0 = blockIdx.x * 32;
  int ntile = blockIdx.y * 4 + wid;     // 0..23
  int n0 = ntile * 16;
  const h16* A = (n0 < DIM) ? qn : kvn;
  v8f acc0 = {}, acc1 = {};
#pragma unroll
  for (int k0 = 0; k0 < DIM; k0 += 32) {
    v16h bfr = load_b_frag(w_in, DIM, n0, k0, lane);
    acc0 = wmma16(load_a_frag(A, DIM, row0,      k0, lane), bfr, acc0);
    acc1 = wmma16(load_a_frag(A, DIM, row0 + 16, k0, lane), bfr, acc1);
  }
  int n_abs = n0 + (lane & 15);
  int mb = (lane >> 4) << 3;
#pragma unroll
  for (int t = 0; t < 2; ++t) {
    v8f acc = t ? acc1 : acc0;
    int rbase = row0 + 16 * t + mb;
#pragma unroll
    for (int j = 0; j < 8; ++j) {
      int tok = rbase + j;
      int bb = tok >> 10, tt = tok & (TSEQ - 1);
      if (n_abs < DIM) {
        int hh = n_abs >> 5, hd = n_abs & 31;
        Qb[(((size_t)(bb * NHH + hh) * TSEQ) + tt) * HDIM + hd] = (h16)acc[j];
      } else if (n_abs < 2 * DIM) {
        int d = n_abs - DIM; int hh = d >> 5, hd = d & 31;
        Kb[(((size_t)(bb * NHH + hh) * TSEQ) + tt) * HDIM + hd] = (h16)acc[j];
      } else {
        int d = n_abs - 2 * DIM; int hh = d >> 5, hd = d & 31;
        Vt[(((size_t)(bb * NHH + hh) * HDIM) + hd) * TSEQ + tt] = (h16)acc[j];
      }
    }
  }
}

// ---------------- flash attention (1 wave / 16-query tile) ----------------
__global__ __launch_bounds__(32) void attn_kernel(
    const h16* __restrict__ Qb, const h16* __restrict__ Kb, const h16* __restrict__ Vt,
    h16* __restrict__ ao)
{
  __shared__ h16 pbuf[16 * 32];
  int lane = threadIdx.x & 31;
  int qt = blockIdx.x, hh = blockIdx.y, bb = blockIdx.z;
  const h16* Q = Qb + (size_t)(bb * NHH + hh) * TSEQ * HDIM;
  const h16* K = Kb + (size_t)(bb * NHH + hh) * TSEQ * HDIM;
  const h16* V = Vt + (size_t)(bb * NHH + hh) * HDIM * TSEQ;
  const float scale = 0.17677669529663687f;  // 1/sqrt(32)
  v16h qa = load_a_frag(Q, HDIM, qt * 16, 0, lane);
  v8f o0 = {}, o1 = {};
  float mrow[8], lrow[8];
#pragma unroll
  for (int j = 0; j < 8; ++j) { mrow[j] = -1e30f; lrow[j] = 0.f; }
  int mb = (lane >> 4) << 3;
  for (int kt = 0; kt < TSEQ; kt += 32) {
    v8f z = {};
    v8f s0 = wmma16(qa, load_b_frag(K, HDIM, kt, 0, lane), z);
    v8f s1 = wmma16(qa, load_b_frag(K, HDIM, kt + 16, 0, lane), z);
#pragma unroll
    for (int j = 0; j < 8; ++j) {
      float a0 = s0[j] * scale, a1 = s1[j] * scale;
      float tm = half_red_max(fmaxf(a0, a1));
      float mnew = fmaxf(mrow[j], tm);
      float cf = __expf(mrow[j] - mnew);
      float p0 = __expf(a0 - mnew), p1 = __expf(a1 - mnew);
      lrow[j] = lrow[j] * cf + half_red_sum(p0 + p1);
      o0[j] *= cf; o1[j] *= cf;
      mrow[j] = mnew;
      pbuf[(mb + j) * 32 + (lane & 15)]      = (h16)p0;
      pbuf[(mb + j) * 32 + 16 + (lane & 15)] = (h16)p1;
    }
    __threadfence_block();
    v16h pa = load_a_frag(pbuf, 32, 0, 0, lane);
    o0 = wmma16(pa, load_b_frag(V, TSEQ, 0, kt, lane), o0);
    o1 = wmma16(pa, load_b_frag(V, TSEQ, 16, kt, lane), o1);
  }
#pragma unroll
  for (int j = 0; j < 8; ++j) {
    int row = qt * 16 + mb + j;
    float invl = 1.0f / lrow[j];
    size_t base = ((size_t)bb * TSEQ + row) * DIM + hh * HDIM;
    ao[base + (lane & 15)]      = (h16)(o0[j] * invl);
    ao[base + 16 + (lane & 15)] = (h16)(o1[j] * invl);
  }
}

// ---------------- attention out-proj + residual (2 M-tiles per wave) ----------------
__global__ __launch_bounds__(128) void outproj_kernel(
    const h16* __restrict__ ao, const h16* __restrict__ w_out,
    const float* __restrict__ resid, float* __restrict__ xb)
{
  int wid = threadIdx.x >> 5, lane = threadIdx.x & 31;
  int row0 = blockIdx.x * 32;
  int n0 = (blockIdx.y * 4 + wid) * 16;
  v8f acc0 = {}, acc1 = {};
#pragma unroll
  for (int k0 = 0; k0 < DIM; k0 += 32) {
    v16h bfr = load_b_frag(w_out, DIM, n0, k0, lane);
    acc0 = wmma16(load_a_frag(ao, DIM, row0,      k0, lane), bfr, acc0);
    acc1 = wmma16(load_a_frag(ao, DIM, row0 + 16, k0, lane), bfr, acc1);
  }
  int n = n0 + (lane & 15);
  int mb = (lane >> 4) << 3;
#pragma unroll
  for (int t = 0; t < 2; ++t) {
    v8f acc = t ? acc1 : acc0;
#pragma unroll
    for (int j = 0; j < 8; ++j) {
      size_t idx = (size_t)(row0 + 16 * t + mb + j) * DIM + n;
      xb[idx] = resid[idx] + acc[j];
    }
  }
}

// ---------------- bridge gated FFN (TDM-staged out weights) ----------------
__global__ __launch_bounds__(128) void bridge_ffn_kernel(
    const h16* __restrict__ hb, const h16* __restrict__ wgate, const h16* __restrict__ wval,
    const h16* __restrict__ wout, const float* __restrict__ xb,
    const float* __restrict__ br_gate, int pair, float* __restrict__ combined)
{
  __shared__ __align__(64) h16 gv[16 * DFF];        // 8 KB
  __shared__ __align__(64) h16 wo_lds[DIM * DFF];   // 64 KB
  int wid = threadIdx.x >> 5, lane = threadIdx.x & 31;
  int row0 = blockIdx.x * 16;
  float sig = 1.0f / (1.0f + __expf(-br_gate[pair]));
#if USE_TDM
  if (threadIdx.x < 32) tdm_load_64k(wout, wo_lds);
#endif
  for (int ft = wid * 4; ft < wid * 4 + 4; ++ft) {
    int n0 = ft * 16;
    v8f ag = {}; v8f av = {};
#pragma unroll
    for (int k0 = 0; k0 < DIM; k0 += 32) {
      v16h a = load_a_frag(hb, DIM, row0, k0, lane);
      ag = wmma16(a, load_b_frag(wgate, DIM, n0, k0, lane), ag);
      av = wmma16(a, load_b_frag(wval, DIM, n0, k0, lane), av);
    }
    int n = n0 + (lane & 15);
    int mb = (lane >> 4) << 3;
#pragma unroll
    for (int j = 0; j < 8; ++j) {
      float gval = ag[j];
      float silu = gval / (1.0f + __expf(-gval));
      gv[(mb + j) * DFF + n] = (h16)(silu * av[j]);
    }
  }
#if USE_TDM
  if (threadIdx.x < 32) __builtin_amdgcn_s_wait_tensorcnt(0);
#endif
  __syncthreads();
#if USE_TDM
  const h16* Bsrc = wo_lds;
#else
  const h16* Bsrc = wout;
#endif
  for (int nt = wid * 2; nt < wid * 2 + 2; ++nt) {
    int n0 = nt * 16;
    v8f acc = {};
#pragma unroll
    for (int k0 = 0; k0 < DFF; k0 += 32)
      acc = wmma16(load_a_frag(gv, DFF, 0, k0, lane),
                   load_b_frag(Bsrc, DFF, n0, k0, lane), acc);
    int n = n0 + (lane & 15);
    int mb = (lane >> 4) << 3;
#pragma unroll
    for (int j = 0; j < 8; ++j) {
      size_t idx = (size_t)(row0 + mb + j) * DIM + n;
      combined[idx] += BW * ((xb[idx] + acc[j]) * sig);
    }
  }
}

// ---------------- final projection (2 M-tiles per wave) ----------------
__global__ __launch_bounds__(128) void final_gemm_kernel(
    const h16* __restrict__ act, const h16* __restrict__ W, float* __restrict__ out)
{
  int wid = threadIdx.x >> 5, lane = threadIdx.x & 31;
  int row0 = blockIdx.x * 32;
  int n0 = (blockIdx.y * 4 + wid) * 16;
  v8f acc0 = {}, acc1 = {};
#pragma unroll
  for (int k0 = 0; k0 < DIM; k0 += 32) {
    v16h bfr = load_b_frag(W, DIM, n0, k0, lane);
    acc0 = wmma16(load_a_frag(act, DIM, row0,      k0, lane), bfr, acc0);
    acc1 = wmma16(load_a_frag(act, DIM, row0 + 16, k0, lane), bfr, acc1);
  }
  int n = n0 + (lane & 15);
  int mb = (lane >> 4) << 3;
#pragma unroll
  for (int t = 0; t < 2; ++t) {
    v8f acc = t ? acc1 : acc0;
#pragma unroll
    for (int j = 0; j < 8; ++j)
      out[(size_t)(row0 + 16 * t + mb + j) * DIM + n] = acc[j];
  }
}

// ---------------- load-balance loss finalize ----------------
__global__ void lb_finalize_kernel(const float* acc, float* out) {
  if (threadIdx.x == 0 && blockIdx.x == 0) {
    float lb = 0.f;
    for (int i = 0; i < 9; ++i) {
      float m = acc[i] * (1.0f / (float)NTOK);
      lb -= m * __logf(m + 1e-8f);
    }
    out[0] = lb * LB_W;
  }
}

// ---------------- host launcher ----------------
extern "C" void kernel_launch(void* const* d_in, const int* in_sizes, int n_in,
                              void* d_out, int out_size, void* d_ws, size_t ws_size,
                              hipStream_t stream) {
  const float* x         = (const float*)d_in[0];
  const float* gr_g      = (const float*)d_in[1];
  const float* gr_b      = (const float*)d_in[2];
  const float* q6_w      = (const float*)d_in[3];
  const float* log_temp  = (const float*)d_in[4];
  const float* grp_g     = (const float*)d_in[5];
  const float* grp_b     = (const float*)d_in[6];
  const float* grp_pw    = (const float*)d_in[7];
  const float* grp_pb    = (const float*)d_in[8];
  const float* me_ng     = (const float*)d_in[9];
  const float* me_nb     = (const float*)d_in[10];
  const float* me_gw     = (const float*)d_in[11];
  const float* me_vw     = (const float*)d_in[12];
  const float* me_ow     = (const float*)d_in[13];
  const float* br_in_w   = (const float*)d_in[14];
  const float* br_out_w  = (const float*)d_in[15];
  const float* br_nq_g   = (const float*)d_in[16];
  const float* br_nq_b   = (const float*)d_in[17];
  const float* br_nkv_g  = (const float*)d_in[18];
  const float* br_nkv_b  = (const float*)d_in[19];
  const float* br_nffn_g = (const float*)d_in[20];
  const float* br_nffn_b = (const float*)d_in[21];
  const float* br_gate_w = (const float*)d_in[22];
  const float* br_val_w  = (const float*)d_in[23];
  const float* br_fout_w = (const float*)d_in[24];
  const float* br_gate   = (const float*)d_in[25];
  const float* out_ng    = (const float*)d_in[26];
  const float* out_nb    = (const float*)d_in[27];
  const float* out_w     = (const float*)d_in[28];
  (void)in_sizes; (void)n_in; (void)out_size; (void)ws_size;

  uint8_t* base = (uint8_t*)d_ws;
  size_t off = 0;
  auto carve = [&](size_t bytes) -> void* {
    void* p = base + off;
    off = (off + bytes + 255) & ~(size_t)255;
    return p;
  };
  h16*   xn_f16   = (h16*)  carve((size_t)NTOK * DIM * 2);
  float* coef     = (float*)carve((size_t)NTOK * 6 * 4);
  float* lb_acc   = (float*)carve(64);
  h16*   e_wg     = (h16*)  carve((size_t)6 * DFF * DIM * 2);
  h16*   e_wv     = (h16*)  carve((size_t)6 * DFF * DIM * 2);
  float* e_bg     = (float*)carve((size_t)6 * DFF * 4);
  float* e_bv     = (float*)carve((size_t)6 * DFF * 4);
  h16*   e_wo     = (h16*)  carve((size_t)6 * DIM * DFF * 2);
  h16*   w_brin   = (h16*)  carve((size_t)3 * 3 * DIM * DIM * 2);
  h16*   w_brout  = (h16*)  carve((size_t)3 * DIM * DIM * 2);
  h16*   w_brg    = (h16*)  carve((size_t)3 * DFF * DIM * 2);
  h16*   w_brv    = (h16*)  carve((size_t)3 * DFF * DIM * 2);
  h16*   w_brf    = (h16*)  carve((size_t)3 * DIM * DFF * 2);
  h16*   w_fin    = (h16*)  carve((size_t)DIM * DIM * 2);
  float* gouts    = (float*)carve((size_t)3 * NTOK * DIM * 4);
  h16*   qn       = (h16*)  carve((size_t)NTOK * DIM * 2);
  h16*   kvn      = (h16*)  carve((size_t)NTOK * DIM * 2);
  h16*   Qb       = (h16*)  carve((size_t)NTOK * DIM * 2);
  h16*   Kb       = (h16*)  carve((size_t)NTOK * DIM * 2);
  h16*   Vt       = (h16*)  carve((size_t)NTOK * DIM * 2);
  h16*   ao       = (h16*)  carve((size_t)NTOK * DIM * 2);
  float* xb       = (float*)carve((size_t)NTOK * DIM * 4);
  h16*   hb       = (h16*)  carve((size_t)NTOK * DIM * 2);
  float* combined = (float*)carve((size_t)NTOK * DIM * 4);

  (void)hipMemsetAsync(lb_acc, 0, 64, stream);

  routing_kernel<<<NTOK / 8, 256, 0, stream>>>(x, gr_g, gr_b, q6_w, log_temp,
                                               grp_g, grp_b, grp_pw, grp_pb,
                                               xn_f16, coef, lb_acc);
  auto cast = [&](const float* s, h16* d, int n) {
    cast_f16_kernel<<<(n + 255) / 256, 256, 0, stream>>>(s, d, n);
  };
  cast(me_ow,     e_wo,    6 * DIM * DFF);
  cast(br_in_w,   w_brin,  3 * 3 * DIM * DIM);
  cast(br_out_w,  w_brout, 3 * DIM * DIM);
  cast(br_gate_w, w_brg,   3 * DFF * DIM);
  cast(br_val_w,  w_brv,   3 * DFF * DIM);
  cast(br_fout_w, w_brf,   3 * DIM * DFF);
  cast(out_w,     w_fin,   DIM * DIM);
  expert_fold_kernel<<<6, 256, 0, stream>>>(me_gw, me_vw, me_ng, me_nb,
                                            e_wg, e_wv, e_bg, e_bv);
  expert_ffn_kernel<<<dim3(NTOK / 16, 3), 128, 0, stream>>>(
      xn_f16, e_wg, e_wv, e_bg, e_bv, e_wo, coef, gouts);
  combine_init_kernel<<<(NTOK * DIM + 255) / 256, 256, 0, stream>>>(gouts, combined);

  const int pair_a[3] = {0, 1, 0};
  const int pair_b[3] = {1, 2, 2};
  for (int i = 0; i < 3; ++i) {
    const float* goa = gouts + (size_t)pair_a[i] * NTOK * DIM;
    const float* gob = gouts + (size_t)pair_b[i] * NTOK * DIM;
    ln_f16_kernel<<<NTOK / 8, 256, 0, stream>>>(goa, br_nq_g + i * DIM, br_nq_b + i * DIM, qn, NTOK);
    ln_f16_kernel<<<NTOK / 8, 256, 0, stream>>>(gob, br_nkv_g + i * DIM, br_nkv_b + i * DIM, kvn, NTOK);
    qkv_kernel<<<dim3(NTOK / 32, 6), 128, 0, stream>>>(
        qn, kvn, w_brin + (size_t)i * 3 * DIM * DIM, Qb, Kb, Vt);
    attn_kernel<<<dim3(TSEQ / 16, NHH, BATCH), 32, 0, stream>>>(Qb, Kb, Vt, ao);
    outproj_kernel<<<dim3(NTOK / 32, 2), 128, 0, stream>>>(
        ao, w_brout + (size_t)i * DIM * DIM, goa, xb);
    ln_f16_kernel<<<NTOK / 8, 256, 0, stream>>>(xb, br_nffn_g + i * DIM, br_nffn_b + i * DIM, hb, NTOK);
    bridge_ffn_kernel<<<NTOK / 16, 128, 0, stream>>>(
        hb, w_brg + (size_t)i * DFF * DIM, w_brv + (size_t)i * DFF * DIM,
        w_brf + (size_t)i * DIM * DFF, xb, br_gate, i, combined);
  }

  ln_f16_kernel<<<NTOK / 8, 256, 0, stream>>>(combined, out_ng, out_nb, hb, NTOK);
  final_gemm_kernel<<<dim3(NTOK / 32, 2), 128, 0, stream>>>(hb, w_fin, (float*)d_out);
  lb_finalize_kernel<<<1, 32, 0, stream>>>(lb_acc, (float*)d_out + (size_t)NTOK * DIM);
}
